// PointConvDensitySetAbstraction_46557445488940
// MI455X (gfx1250) — matrix operand order repaired and checked by reading, not compiled, not run on hardware
//
#include <hip/hip_runtime.h>

typedef __attribute__((ext_vector_type(16))) _Float16 v16h;
typedef __attribute__((ext_vector_type(8)))  float    v8f;

#define Bc 4
#define Nc 4096
#define Sc 1024
#define Kc 32
static constexpr long long COLS = (long long)Bc * Sc * Kc; // 131072
#define EPSc 1e-5f

// ---------------------------------------------------------------------------
// Packed-tile index helpers (512 halves per 16x32 A tile / 32x16 B tile),
// for v_wmma_f32_16x16x32_f16, k = 16*(lane/16) + element in both A and B (contiguous).
// ---------------------------------------------------------------------------
__device__ inline size_t packA_idx(int m, int k, int Kblocks) {
  int Kb = k & 31;
  int lane = ((Kb >> 4) << 4) + (m & 15);
  int h = Kb & 15;
  return ((size_t)(m >> 4) * Kblocks + (k >> 5)) * 512 + lane * 16 + h;
}
__device__ inline size_t packB_idx(int k, long long col, int Kblocks) {
  int Kb = k & 31;
  int lane = ((Kb >> 4) << 4) + (int)(col & 15);
  int h = Kb & 15;
  return ((size_t)(col >> 4) * Kblocks + (k >> 5)) * 512 + lane * 16 + h;
}

// ---------------------------------------------------------------------------
// GEMM: Y[Cout][cols] = A(packed)[Cout/16][Kb][512] * B(packed)[cols/16][Kb][512]
// 8 waves/block; each wave computes a 16 x 64 output strip (4 col tiles),
// reusing one A tile per k-step across the 4 WMMAs.
// ---------------------------------------------------------------------------
__global__ void gemm_wmma_kernel(const _Float16* __restrict__ Ap,
                                 const _Float16* __restrict__ Bp,
                                 float* __restrict__ Y,
                                 int Kblocks, long long cols) {
  int lane = threadIdx.x & 31;
  int wave = threadIdx.x >> 5;
  int m0 = blockIdx.y * 16;
  int colTile0 = (blockIdx.x * 8 + wave) * 4;

  const _Float16* ap = Ap + ((size_t)blockIdx.y * Kblocks) * 512 + lane * 16;
  const _Float16* bp0 = Bp + ((size_t)(colTile0 + 0) * Kblocks) * 512 + lane * 16;
  const _Float16* bp1 = Bp + ((size_t)(colTile0 + 1) * Kblocks) * 512 + lane * 16;
  const _Float16* bp2 = Bp + ((size_t)(colTile0 + 2) * Kblocks) * 512 + lane * 16;
  const _Float16* bp3 = Bp + ((size_t)(colTile0 + 3) * Kblocks) * 512 + lane * 16;

  v8f c0 = {}, c1 = {}, c2 = {}, c3 = {};
  for (int kb = 0; kb < Kblocks; ++kb) {
    __builtin_prefetch((const void*)(ap + (kb + 2) * 512), 0, 0);
    __builtin_prefetch((const void*)(bp0 + (kb + 2) * 512), 0, 0);
    v16h a = *(const v16h*)(ap + (size_t)kb * 512);
    v16h b0 = *(const v16h*)(bp0 + (size_t)kb * 512);
    c0 = __builtin_amdgcn_wmma_f32_16x16x32_f16(false, a, false, b0, (short)0, c0, false, false);
    v16h b1 = *(const v16h*)(bp1 + (size_t)kb * 512);
    c1 = __builtin_amdgcn_wmma_f32_16x16x32_f16(false, a, false, b1, (short)0, c1, false, false);
    v16h b2 = *(const v16h*)(bp2 + (size_t)kb * 512);
    c2 = __builtin_amdgcn_wmma_f32_16x16x32_f16(false, a, false, b2, (short)0, c2, false, false);
    v16h b3 = *(const v16h*)(bp3 + (size_t)kb * 512);
    c3 = __builtin_amdgcn_wmma_f32_16x16x32_f16(false, a, false, b3, (short)0, c3, false, false);
  }
  int N = lane & 15;
  int hi = (lane >> 4) & 1;
#pragma unroll
  for (int r = 0; r < 8; ++r) {
    int M = m0 + r + hi * 8;
    long long base = (long long)M * cols + (long long)colTile0 * 16 + N;
    Y[base + 0]  = c0[r];
    Y[base + 16] = c1[r];
    Y[base + 32] = c2[r];
    Y[base + 48] = c3[r];
  }
}

// ---------------------------------------------------------------------------
// Aggregation: nf[b,s,d,c] = sum_k x[d][bs*32+k] * w[c][bs*32+k]; K==32 exactly.
// A pre-packed per (bs, d-tile); B pre-packed per bs. Epilogue scatters straight
// into the packed-B layout consumed by the final linear GEMM (Kblocks=64).
// ---------------------------------------------------------------------------
__global__ void agg_wmma_kernel(const _Float16* __restrict__ Ap,  // [bs][8][512]
                                const _Float16* __restrict__ Bp,  // [bs][512]
                                _Float16* __restrict__ nfBp) {    // packed B, K=2048, cols=4096
  int lane = threadIdx.x & 31;
  int wave = threadIdx.x >> 5;
  int t = blockIdx.x * 4 + wave;   // 0..32767
  int bs = t >> 3;
  int mt = t & 7;
  v16h a = *(const v16h*)(Ap + ((size_t)bs * 8 + mt) * 512 + lane * 16);
  v16h b = *(const v16h*)(Bp + (size_t)bs * 512 + lane * 16);
  v8f c = {};
  c = __builtin_amdgcn_wmma_f32_16x16x32_f16(false, a, false, b, (short)0, c, false, false);
  int N = lane & 15;
  int hi = (lane >> 4) & 1;
#pragma unroll
  for (int r = 0; r < 8; ++r) {
    int f = (mt * 16 + r + hi * 8) * 16 + N;   // feature index 0..2047
    nfBp[packB_idx(f, bs, 64)] = (_Float16)c[r];
  }
}

// ---------------------------------------------------------------------------
// Density: inv_density[b][n] = 1 / mean_m( 4 * exp(-50*dist(n,m)) )
// ---------------------------------------------------------------------------
__global__ void density_kernel(const float* __restrict__ xyz,
                               float* __restrict__ inv_den) {
  int b = blockIdx.y;
  int n = blockIdx.x * 256 + threadIdx.x;
  const float* P = xyz + (long long)b * Nc * 3;
  float px = P[n * 3 + 0], py = P[n * 3 + 1], pz = P[n * 3 + 2];
  __shared__ float sx[256], sy[256], sz[256];
  float acc = 0.f;
  for (int m0 = 0; m0 < Nc; m0 += 256) {
    int m = m0 + threadIdx.x;
    sx[threadIdx.x] = P[m * 3 + 0];
    sy[threadIdx.x] = P[m * 3 + 1];
    sz[threadIdx.x] = P[m * 3 + 2];
    __syncthreads();
    for (int j = 0; j < 256; ++j) {
      float dx = px - sx[j], dy = py - sy[j], dz = pz - sz[j];
      float d2 = dx * dx + dy * dy + dz * dz;
      float d = sqrtf(fmaxf(d2, 0.f));
      acc += expf(-50.f * d);
    }
    __syncthreads();
  }
  float density = acc * (4.0f / (float)Nc);
  inv_den[b * Nc + n] = 1.0f / density;
}

// ---------------------------------------------------------------------------
// Farthest point sampling: one block per batch, 256 threads x 16 pts each.
// ---------------------------------------------------------------------------
__global__ void fps_kernel(const float* __restrict__ xyz, int* __restrict__ fidx) {
  int b = blockIdx.x;
  const float* P = xyz + (long long)b * Nc * 3;
  int t = threadIdx.x;
  float mind[16], x[16], y[16], z[16];
#pragma unroll
  for (int j = 0; j < 16; ++j) {
    int n = t + j * 256;
    mind[j] = 1e10f;
    x[j] = P[n * 3 + 0];
    y[j] = P[n * 3 + 1];
    z[j] = P[n * 3 + 2];
  }
  __shared__ float sv[256];
  __shared__ int si[256];
  __shared__ float lp[3];
  int last = 0;
  for (int i = 0; i < Sc; ++i) {
    if (t == 0) {
      fidx[b * Sc + i] = last;
      lp[0] = P[last * 3 + 0];
      lp[1] = P[last * 3 + 1];
      lp[2] = P[last * 3 + 2];
    }
    __syncthreads();
    float lx = lp[0], ly = lp[1], lz = lp[2];
    float bv = -1.f;
    int bi = 0x7fffffff;
#pragma unroll
    for (int j = 0; j < 16; ++j) {
      float dx = x[j] - lx, dy = y[j] - ly, dz = z[j] - lz;
      float d = dx * dx + dy * dy + dz * dz;
      mind[j] = fminf(mind[j], d);
      int n = t + j * 256;
      if (mind[j] > bv || (mind[j] == bv && n < bi)) { bv = mind[j]; bi = n; }
    }
    sv[t] = bv; si[t] = bi;
    __syncthreads();
    for (int off = 128; off > 0; off >>= 1) {
      if (t < off) {
        float ov = sv[t + off]; int oi = si[t + off];
        if (ov > sv[t] || (ov == sv[t] && oi < si[t])) { sv[t] = ov; si[t] = oi; }
      }
      __syncthreads();
    }
    last = si[0];
    __syncthreads();
  }
}

// ---------------------------------------------------------------------------
// kNN (top-32 smallest d2) + emit fps_xyz into d_out
// ---------------------------------------------------------------------------
__global__ void knn_kernel(const float* __restrict__ xyz,
                           const int* __restrict__ fidx,
                           int* __restrict__ nidx,
                           float* __restrict__ fps_out) {
  int tid = blockIdx.x * 256 + threadIdx.x;   // 0..4095
  int b = tid / Sc, s = tid % Sc;
  const float* P = xyz + (long long)b * Nc * 3;
  int f = fidx[b * Sc + s];
  float px = P[f * 3 + 0], py = P[f * 3 + 1], pz = P[f * 3 + 2];
  fps_out[tid * 3 + 0] = px;
  fps_out[tid * 3 + 1] = py;
  fps_out[tid * 3 + 2] = pz;
  float dist[Kc];
  int idx[Kc];
  for (int k = 0; k < Kc; ++k) { dist[k] = 3.4e38f; idx[k] = 0; }
  float worst = 3.4e38f;
  int wslot = 0;
  for (int n = 0; n < Nc; ++n) {
    float dx = px - P[n * 3 + 0], dy = py - P[n * 3 + 1], dz = pz - P[n * 3 + 2];
    float d2 = dx * dx + dy * dy + dz * dz;
    if (d2 < worst) {
      dist[wslot] = d2;
      idx[wslot] = n;
      worst = dist[0]; wslot = 0;
      for (int k = 1; k < Kc; ++k)
        if (dist[k] >= worst) { worst = dist[k]; wslot = k; }
    }
  }
  for (int k = 0; k < Kc; ++k) nidx[(long long)tid * Kc + k] = idx[k];
}

// ---------------------------------------------------------------------------
// Grouping: gxyz (f32 [3][COLS]), X0 packed-B f16 (K=96, Kblocks=3), ds_raw
// ---------------------------------------------------------------------------
__global__ void group_kernel(const float* __restrict__ xyz,
                             const float* __restrict__ feat,
                             const int* __restrict__ nidx,
                             const float* __restrict__ inv_den,
                             const float* __restrict__ fps_xyz,
                             float* __restrict__ gxyz,
                             _Float16* __restrict__ X0p,
                             float* __restrict__ ds_raw) {
  long long col = (long long)blockIdx.x * 256 + threadIdx.x; // 0..COLS-1
  int bs = (int)(col >> 5);
  int b = bs >> 10;
  int n = nidx[col];
  const float* P = xyz + ((long long)b * Nc + n) * 3;
  float gx = P[0] - fps_xyz[bs * 3 + 0];
  float gy = P[1] - fps_xyz[bs * 3 + 1];
  float gz = P[2] - fps_xyz[bs * 3 + 2];
  gxyz[0 * COLS + col] = gx;
  gxyz[1 * COLS + col] = gy;
  gxyz[2 * COLS + col] = gz;
  X0p[packB_idx(0, col, 3)] = (_Float16)gx;
  X0p[packB_idx(1, col, 3)] = (_Float16)gy;
  X0p[packB_idx(2, col, 3)] = (_Float16)gz;
  const float* F = feat + ((long long)b * Nc + n) * 64;
  for (int c2 = 0; c2 < 64; ++c2)
    X0p[packB_idx(3 + c2, col, 3)] = (_Float16)F[c2];
  for (int c2 = 67; c2 < 96; ++c2)
    X0p[packB_idx(c2, col, 3)] = (_Float16)0.f;
  ds_raw[col] = inv_den[b * Nc + n];
}

__global__ void dsnorm_kernel(const float* __restrict__ ds_raw,
                              float* __restrict__ dsn) {
  int bs = blockIdx.x * 256 + threadIdx.x;  // 0..4095
  long long base = (long long)bs * Kc;
  float mx = -3.4e38f;
  for (int k = 0; k < Kc; ++k) mx = fmaxf(mx, ds_raw[base + k]);
  float inv = 1.0f / mx;
  for (int k = 0; k < Kc; ++k) dsn[base + k] = ds_raw[base + k] * inv;
}

// ---------------------------------------------------------------------------
// Small 1x1 conv (VALU): Y[Cout][cols] = W[Cout][Cin] * X[Cin][cols]
// ---------------------------------------------------------------------------
__global__ void conv_small_kernel(const float* __restrict__ W,
                                  const float* __restrict__ X,
                                  float* __restrict__ Y,
                                  int Cout, int Cin, long long cols) {
  long long col = (long long)blockIdx.x * 256 + threadIdx.x;
  if (col >= cols) return;
  for (int o = 0; o < Cout; ++o) {
    float acc = 0.f;
    for (int ci = 0; ci < Cin; ++ci)
      acc += W[o * Cin + ci] * X[(long long)ci * cols + col];
    Y[(long long)o * cols + col] = acc;
  }
}

// ---------------------------------------------------------------------------
// Per-channel mean/var (population) over cols; one block per channel
// ---------------------------------------------------------------------------
__global__ void stats_kernel(const float* __restrict__ Y,
                             float* __restrict__ mean, float* __restrict__ var,
                             long long cols) {
  int c = blockIdx.x;
  const float* row = Y + (long long)c * cols;
  int t = threadIdx.x;
  float s = 0.f, s2 = 0.f;
  for (long long i = t; i < cols; i += 256) {
    float v = row[i];
    s += v; s2 += v * v;
  }
  __shared__ float ss[256], sq[256];
  ss[t] = s; sq[t] = s2;
  __syncthreads();
  for (int off = 128; off > 0; off >>= 1) {
    if (t < off) { ss[t] += ss[t + off]; sq[t] += sq[t + off]; }
    __syncthreads();
  }
  if (t == 0) {
    float m = ss[0] / (float)cols;
    mean[c] = m;
    var[c] = sq[0] / (float)cols - m * m;
  }
}

// ---------------------------------------------------------------------------
// BN + ReLU apply variants
// ---------------------------------------------------------------------------
// -> packed-B f16 (feeds next WMMA GEMM)
__global__ void bn_relu_packB_kernel(const float* __restrict__ Y,
                                     const float* __restrict__ g,
                                     const float* __restrict__ be,
                                     const float* __restrict__ mean,
                                     const float* __restrict__ var,
                                     _Float16* __restrict__ Bp,
                                     long long cols, int Kblocks) {
  long long i = (long long)blockIdx.x * 256 + threadIdx.x;
  int c = (int)(i / cols);
  long long col = i % cols;
  float val = g[c] * (Y[i] - mean[c]) * rsqrtf(var[c] + EPSc) + be[c];
  Bp[packB_idx(c, col, Kblocks)] = (_Float16)fmaxf(val, 0.f);
}

// -> plain channel-major f16 (layer-3 output, consumed by mulds)
__global__ void bn_relu_f16_kernel(const float* __restrict__ Y,
                                   const float* __restrict__ g,
                                   const float* __restrict__ be,
                                   const float* __restrict__ mean,
                                   const float* __restrict__ var,
                                   _Float16* __restrict__ Xo, long long cols) {
  long long i = (long long)blockIdx.x * 256 + threadIdx.x;
  int c = (int)(i / cols);
  float val = g[c] * (Y[i] - mean[c]) * rsqrtf(var[c] + EPSc) + be[c];
  Xo[i] = (_Float16)fmaxf(val, 0.f);
}

// -> plain f32
__global__ void bn_relu_f32_kernel(const float* __restrict__ Y,
                                   const float* __restrict__ g,
                                   const float* __restrict__ be,
                                   const float* __restrict__ mean,
                                   const float* __restrict__ var,
                                   float* __restrict__ Xo, long long cols) {
  long long i = (long long)blockIdx.x * 256 + threadIdx.x;
  int c = (int)(i / cols);
  float val = g[c] * (Y[i] - mean[c]) * rsqrtf(var[c] + EPSc) + be[c];
  Xo[i] = fmaxf(val, 0.f);
}

// -> aggregation B tiles: element (channel c = N, col -> bs,K)
__global__ void bn_relu_packAggB_kernel(const float* __restrict__ Y,
                                        const float* __restrict__ g,
                                        const float* __restrict__ be,
                                        const float* __restrict__ mean,
                                        const float* __restrict__ var,
                                        _Float16* __restrict__ Bp,   // [bs][512]
                                        long long cols) {
  long long i = (long long)blockIdx.x * 256 + threadIdx.x;   // 16*cols
  int c = (int)(i / cols);
  long long col = i % cols;
  int bs = (int)(col >> 5);
  int K = (int)(col & 31);
  float val = g[c] * (Y[i] - mean[c]) * rsqrtf(var[c] + EPSc) + be[c];
  int lane = ((K >> 4) << 4) + c;
  Bp[(size_t)bs * 512 + lane * 16 + (K & 15)] = (_Float16)fmaxf(val, 0.f);
}

// Final BN1d + relu; Ylin is [128][4096], out is [bs][128]
__global__ void bn_relu_final_kernel(const float* __restrict__ Ylin,
                                     const float* __restrict__ g,
                                     const float* __restrict__ be,
                                     const float* __restrict__ mean,
                                     const float* __restrict__ var,
                                     float* __restrict__ out) {
  int i = blockIdx.x * 256 + threadIdx.x;   // 0..524287
  int o = i >> 12;          // /4096
  int bs = i & 4095;
  float val = g[o] * (Ylin[i] - mean[o]) * rsqrtf(var[o] + EPSc) + be[o];
  out[(long long)bs * 128 + o] = fmaxf(val, 0.f);
}

// f32 weights -> packed-A f16 tiles (zero-padded K)
__global__ void padw_packA_kernel(const float* __restrict__ W, _Float16* __restrict__ Ap,
                                  int Cout, int Cin, int Kblocks) {
  int Kpad = Kblocks * 32;
  int i = blockIdx.x * 256 + threadIdx.x;
  if (i >= Cout * Kpad) return;
  int o = i / Kpad, k = i % Kpad;
  float v = (k < Cin) ? W[o * Cin + k] : 0.f;
  Ap[packA_idx(o, k, Kblocks)] = (_Float16)v;
}

// A_agg (packed per (bs, d-tile)) = f16( X3[d][col] * ds[col] )
__global__ void mulds_packA_kernel(const _Float16* __restrict__ X3,
                                   const float* __restrict__ ds,
                                   _Float16* __restrict__ Ap, long long cols) {
  long long i = (long long)blockIdx.x * 256 + threadIdx.x; // 128*cols
  int d = (int)(i / cols);
  long long col = i % cols;
  int bs = (int)(col >> 5);
  int Kb = (int)(col & 31);
  int lane = ((Kb >> 4) << 4) + (d & 15);
  int h = Kb & 15;
  float v = (float)X3[i] * ds[col];
  Ap[((size_t)bs * 8 + (d >> 4)) * 512 + lane * 16 + h] = (_Float16)v;
}

// ---------------------------------------------------------------------------
// Host launcher
// ---------------------------------------------------------------------------
extern "C" void kernel_launch(void* const* d_in, const int* in_sizes, int n_in,
                              void* d_out, int out_size, void* d_ws, size_t ws_size,
                              hipStream_t stream) {
  (void)in_sizes; (void)n_in; (void)out_size; (void)ws_size;
  const float* xyz  = (const float*)d_in[0];
  const float* feat = (const float*)d_in[1];
  const float* mW0 = (const float*)d_in[2];
  const float* mG0 = (const float*)d_in[4];
  const float* mB0 = (const float*)d_in[5];
  const float* mW1 = (const float*)d_in[6];
  const float* mG1 = (const float*)d_in[8];
  const float* mB1 = (const float*)d_in[9];
  const float* mW2 = (const float*)d_in[10];
  const float* mG2 = (const float*)d_in[12];
  const float* mB2 = (const float*)d_in[13];
  const float* dW0 = (const float*)d_in[14];
  const float* dG0 = (const float*)d_in[16];
  const float* dB0 = (const float*)d_in[17];
  const float* dW1 = (const float*)d_in[18];
  const float* dG1 = (const float*)d_in[20];
  const float* dB1 = (const float*)d_in[21];
  const float* dW2 = (const float*)d_in[22];
  const float* dG2 = (const float*)d_in[24];
  const float* dB2 = (const float*)d_in[25];
  const float* wW0 = (const float*)d_in[26];
  const float* wG0 = (const float*)d_in[28];
  const float* wB0 = (const float*)d_in[29];
  const float* wW1 = (const float*)d_in[30];
  const float* wG1 = (const float*)d_in[32];
  const float* wB1 = (const float*)d_in[33];
  const float* wW2 = (const float*)d_in[34];
  const float* wG2 = (const float*)d_in[36];
  const float* wB2 = (const float*)d_in[37];
  const float* linW = (const float*)d_in[38];
  const float* bnlG = (const float*)d_in[40];
  const float* bnlB = (const float*)d_in[41];

  float* out_fps = (float*)d_out;                    // (B,S,3) = 12288 f32
  float* out_nf  = (float*)d_out + (size_t)Bc * Sc * 3;

  // ---- workspace layout ----
  char* W = (char*)d_ws;
  size_t off = 0;
  float*    bufY  = (float*)   (W + off); off += (size_t)128 * COLS * 4; // 64MB
  _Float16* bufA  = (_Float16*)(W + off); off += (size_t)128 * COLS * 2; // 32MB
  _Float16* bufB  = (_Float16*)(W + off); off += (size_t)128 * COLS * 2; // 32MB
  float*    gxyz  = (float*)   (W + off); off += (size_t)3 * COLS * 4;
  float*    dsraw = (float*)   (W + off); off += (size_t)COLS * 4;
  float*    dsn   = (float*)   (W + off); off += (size_t)COLS * 4;
  float*    dsfin = (float*)   (W + off); off += (size_t)COLS * 4;
  float*    R1    = (float*)   (W + off); off += (size_t)16 * COLS * 4;
  float*    R2    = (float*)   (W + off); off += (size_t)16 * COLS * 4;
  _Float16* w16   = (_Float16*)(W + off); off += (size_t)16 * COLS * 2;  // 4096x512 agg-B tiles
  _Float16* Wp    = (_Float16*)(W + off); off += (size_t)128 * 2048 * 2;
  float*    invd  = (float*)   (W + off); off += (size_t)Bc * Nc * 4;
  float*    meanp = (float*)   (W + off); off += 128 * 4;
  float*    varp  = (float*)   (W + off); off += 128 * 4;
  int*      nidx  = (int*)     (W + off); off += (size_t)COLS * 4;
  int*      fidx  = (int*)     (W + off); off += (size_t)Bc * Sc * 4;

  _Float16* nfBp = (_Float16*)bufY;       // packed B for linear (256x64x512 halves)
  float*    Ylin = (float*)bufB;          // 128 x 4096 f32 (reuse after X3 consumed)

  // ---- 1) density / 2) FPS / 3) kNN / 4) grouping ----
  density_kernel<<<dim3(Nc / 256, Bc), 256, 0, stream>>>(xyz, invd);
  fps_kernel<<<Bc, 256, 0, stream>>>(xyz, fidx);
  knn_kernel<<<(Bc * Sc) / 256, 256, 0, stream>>>(xyz, fidx, nidx, out_fps);
  group_kernel<<<(int)(COLS / 256), 256, 0, stream>>>(xyz, feat, nidx, invd,
                                                      out_fps, gxyz, bufA, dsraw);
  dsnorm_kernel<<<(Bc * Sc) / 256, 256, 0, stream>>>(dsraw, dsn);

  const int cb = (int)(COLS / 256);
  // ---- 5) weight net: 3->8->8->16 (VALU); last apply packs agg-B tiles ----
  conv_small_kernel<<<cb, 256, 0, stream>>>(wW0, gxyz, R1, 8, 3, COLS);
  stats_kernel<<<8, 256, 0, stream>>>(R1, meanp, varp, COLS);
  bn_relu_f32_kernel<<<(int)(8 * COLS / 256), 256, 0, stream>>>(R1, wG0, wB0, meanp, varp, R2, COLS);
  conv_small_kernel<<<cb, 256, 0, stream>>>(wW1, R2, R1, 8, 8, COLS);
  stats_kernel<<<8, 256, 0, stream>>>(R1, meanp, varp, COLS);
  bn_relu_f32_kernel<<<(int)(8 * COLS / 256), 256, 0, stream>>>(R1, wG1, wB1, meanp, varp, R2, COLS);
  conv_small_kernel<<<cb, 256, 0, stream>>>(wW2, R2, R1, 16, 8, COLS);
  stats_kernel<<<16, 256, 0, stream>>>(R1, meanp, varp, COLS);
  bn_relu_packAggB_kernel<<<(int)(16 * COLS / 256), 256, 0, stream>>>(R1, wG2, wB2, meanp, varp, w16, COLS);

  // ---- 6) density net: 1->16->8->1 (VALU) ----
  conv_small_kernel<<<cb, 256, 0, stream>>>(dW0, dsn, R1, 16, 1, COLS);
  stats_kernel<<<16, 256, 0, stream>>>(R1, meanp, varp, COLS);
  bn_relu_f32_kernel<<<(int)(16 * COLS / 256), 256, 0, stream>>>(R1, dG0, dB0, meanp, varp, R2, COLS);
  conv_small_kernel<<<cb, 256, 0, stream>>>(dW1, R2, R1, 8, 16, COLS);
  stats_kernel<<<8, 256, 0, stream>>>(R1, meanp, varp, COLS);
  bn_relu_f32_kernel<<<(int)(8 * COLS / 256), 256, 0, stream>>>(R1, dG1, dB1, meanp, varp, R2, COLS);
  conv_small_kernel<<<cb, 256, 0, stream>>>(dW2, R2, R1, 1, 8, COLS);
  stats_kernel<<<1, 256, 0, stream>>>(R1, meanp, varp, COLS);
  bn_relu_f32_kernel<<<(int)(COLS / 256), 256, 0, stream>>>(R1, dG2, dB2, meanp, varp, dsfin, COLS);

  // ---- 7) point MLP via WMMA: 67(->96)->64->64->128 ----
  padw_packA_kernel<<<(64 * 96 + 255) / 256, 256, 0, stream>>>(mW0, Wp, 64, 67, 3);
  gemm_wmma_kernel<<<dim3(256, 4), 256, 0, stream>>>(Wp, bufA, bufY, 3, COLS);
  stats_kernel<<<64, 256, 0, stream>>>(bufY, meanp, varp, COLS);
  bn_relu_packB_kernel<<<(int)(64 * COLS / 256), 256, 0, stream>>>(bufY, mG0, mB0, meanp, varp, bufB, COLS, 2);

  padw_packA_kernel<<<(64 * 64 + 255) / 256, 256, 0, stream>>>(mW1, Wp, 64, 64, 2);
  gemm_wmma_kernel<<<dim3(256, 4), 256, 0, stream>>>(Wp, bufB, bufY, 2, COLS);
  stats_kernel<<<64, 256, 0, stream>>>(bufY, meanp, varp, COLS);
  bn_relu_packB_kernel<<<(int)(64 * COLS / 256), 256, 0, stream>>>(bufY, mG1, mB1, meanp, varp, bufA, COLS, 2);

  padw_packA_kernel<<<(128 * 64 + 255) / 256, 256, 0, stream>>>(mW2, Wp, 128, 64, 2);
  gemm_wmma_kernel<<<dim3(256, 8), 256, 0, stream>>>(Wp, bufA, bufY, 2, COLS);
  stats_kernel<<<128, 256, 0, stream>>>(bufY, meanp, varp, COLS);
  bn_relu_f16_kernel<<<(int)(128 * COLS / 256), 256, 0, stream>>>(bufY, mG2, mB2, meanp, varp, bufB, COLS);

  // ---- 8) x = x * ds  (packs aggregation A tiles) ----
  mulds_packA_kernel<<<(int)(128 * COLS / 256), 256, 0, stream>>>(bufB, dsfin, bufA, COLS);

  // ---- 9) weighted aggregation via WMMA; writes packed-B for the linear ----
  agg_wmma_kernel<<<8192, 128, 0, stream>>>(bufA, w16, nfBp);

  // ---- 10) linear 2048->128 via WMMA + BN1d + relu ----
  padw_packA_kernel<<<(128 * 2048 + 255) / 256, 256, 0, stream>>>(linW, Wp, 128, 2048, 64);
  gemm_wmma_kernel<<<dim3(8, 8), 256, 0, stream>>>(Wp, nfBp, Ylin, 64, 4096);
  stats_kernel<<<128, 256, 0, stream>>>(Ylin, meanp, varp, 4096);
  bn_relu_final_kernel<<<(128 * 4096) / 256, 256, 0, stream>>>(Ylin, bnlG, bnlB, meanp, varp, out_nf);
}
